// GDBlock_71605694759231
// MI455X (gfx1250) — matrix operand order repaired and checked by reading, not compiled, not run on hardware
//
#include <hip/hip_runtime.h>

#define B_   2
#define S_   2048
#define E_   768
#define H_   12
#define DFF_ 3072
#define V_   32000
#define ROWS_ 4096
#define LN_EPS 1e-5f

typedef __bf16 bf16_t;
typedef __attribute__((ext_vector_type(16))) __bf16 bf16x16;
typedef __attribute__((ext_vector_type(8)))  __bf16 bf16x8;
typedef __attribute__((ext_vector_type(8)))  float  f32x8;
typedef __attribute__((ext_vector_type(4)))  int    v4i;

static __device__ inline bf16_t f2bf(float f){
  unsigned u = __builtin_bit_cast(unsigned, f);
  unsigned r = u + 0x7FFFu + ((u >> 16) & 1u);   // round-to-nearest-even
  return __builtin_bit_cast(bf16_t, (unsigned short)(r >> 16));
}

static __device__ inline f32x8 wmma_bf16(bf16x16 a, bf16x16 b, f32x8 c){
  return __builtin_amdgcn_wmma_f32_16x16x32_bf16(false, a, false, b, (short)0, c, false, false);
}

// ---- async global->LDS 16B copy (CDNA5), guarded with fallback ------------
#if defined(__gfx1250__) && __has_builtin(__builtin_amdgcn_global_load_async_to_lds_b128)
#define HAVE_ASYNC_LDS 1
typedef __attribute__((address_space(3))) v4i* lds_v4i_ptr;
static __device__ inline void async_cp16(const void* g, void* l){
  __builtin_amdgcn_global_load_async_to_lds_b128(
      (v4i*)g, (lds_v4i_ptr)(unsigned)(uintptr_t)l, 0, 0);
}
#else
static __device__ inline void async_cp16(const void* g, void* l){
  *(bf16x8*)l = *(const bf16x8*)g;
}
#endif

static __device__ inline void wait_async(){
#if defined(__gfx1250__) && __has_builtin(__builtin_amdgcn_s_wait_asynccnt)
  __builtin_amdgcn_s_wait_asynccnt(0);
#elif defined(HAVE_ASYNC_LDS)
  asm volatile("s_wait_asynccnt 0x0" ::: "memory");
#endif
}

// A operand: 16x32 tile from LDS stored row-major [row][k] (pitch in elements).
static __device__ inline bf16x16 load_A(const bf16_t* base, int pitch, int row0, int k0, int lane){
  const int m = lane & 15, half = lane >> 4;
  const bf16_t* p = base + (row0 + m) * pitch + k0 + half * 8;
  bf16x16 r;
  *(bf16x8*)&r       = *(const bf16x8*)p;
  *((bf16x8*)&r + 1) = *(const bf16x8*)(p + 16);
  return r;
}

// B operand: 32x16 (KxN) tile from LDS stored [n][k] (pitch in elements).
static __device__ inline bf16x16 load_B(const bf16_t* base, int pitch, int n0, int k0, int lane){
  const int n = lane & 15, half = lane >> 4;
  const bf16_t* p = base + (n0 + n) * pitch + k0 + half * 8;
  bf16x16 r;
  *(bf16x8*)&r       = *(const bf16x8*)p;
  *((bf16x8*)&r + 1) = *(const bf16x8*)(p + 16);
  return r;
}

// ---------------------------------------------------------------------------
// K0: f32 -> bf16 weight conversion (one pass; removes VALU converts from hot
// loops and makes all weight staging pure copies -> async-to-LDS eligible)
// ---------------------------------------------------------------------------
__global__ __launch_bounds__(256, 1)
void k_cvt_bf16(const float* __restrict__ src, bf16_t* __restrict__ dst, int n8){
  int i = blockIdx.x * 256 + threadIdx.x;
  if (i >= n8) return;
  const float4* s = (const float4*)src + (size_t)i * 2;
  float4 x = s[0], y = s[1];
  bf16x8 o = { f2bf(x.x), f2bf(x.y), f2bf(x.z), f2bf(x.w),
               f2bf(y.x), f2bf(y.y), f2bf(y.z), f2bf(y.w) };
  *(bf16x8*)(dst + (size_t)i * 8) = o;
}

// ---------------------------------------------------------------------------
// K1: flash-style kernelized expectation over the vocab (bf16 W_e).
// ---------------------------------------------------------------------------
__global__ __launch_bounds__(512, 1)
void k_vocab_expect(const float* __restrict__ f_k, const bf16_t* __restrict__ W_e,
                    float* __restrict__ num, float* __restrict__ den){
  extern __shared__ char smem[];
  bf16_t* fk_sh  = (bf16_t*)smem;                 // [64][768] bf16   96KB
  bf16_t* Wrow   = fk_sh + 64*E_;                 // [32][768] bf16   48KB
  bf16_t* Wt     = Wrow + 32*E_;                  // [768][32] bf16   48KB
  float*  Plog   = (float*)(Wt + E_*32);          // [4][64][32] f32  32KB
  bf16_t* Pb     = (bf16_t*)(Plog + 4*64*32);     // [64][32] bf16     4KB
  float*  den_sh = (float*)(Pb + 64*32);          // [64] f32

  const int tid = threadIdx.x;
  const int lane = tid & 31, wave = tid >> 5;
  const int r = wave >> 2, c = wave & 3;
  const int row0g = blockIdx.x * 64;
  const int vbase = blockIdx.y * (V_/4);

  for (int i = tid; i < 64*E_; i += 512)
    fk_sh[i] = f2bf(f_k[(size_t)(row0g + (i/E_))*E_ + (i%E_)]);
  if (tid < 64) den_sh[tid] = 0.f;

  const f32x8 vz = {0,0,0,0,0,0,0,0};
  f32x8 acc[12];
  for (int t = 0; t < 12; ++t) acc[t] = vz;

  for (int vt = 0; vt < V_/4; vt += 32){
    __syncthreads();
    // row-major copy: pure 16B async copies
    for (int i8 = tid; i8 < 32*96; i8 += 512){
      int v = i8 / 96, k8 = (i8 % 96) * 8;
      async_cp16(&W_e[(size_t)(vbase + vt + v)*E_ + k8], &Wrow[v*E_ + k8]);
    }
    // transposed copy (register transpose, b64 reads)
    for (int i4 = tid; i4 < 32*192; i4 += 512){
      int v = i4 / 192, k0 = (i4 % 192) * 4;
      bf16_t tmp[4];
      *(uint2*)tmp = *(const uint2*)&W_e[(size_t)(vbase + vt + v)*E_ + k0];
      Wt[(k0+0)*32 + v] = tmp[0];
      Wt[(k0+1)*32 + v] = tmp[1];
      Wt[(k0+2)*32 + v] = tmp[2];
      Wt[(k0+3)*32 + v] = tmp[3];
    }
    wait_async();
    __syncthreads();
    // GEMM1: partial logits over K chunk [c*192, c*192+192)
    {
      f32x8 d0 = vz, d1 = vz;
      for (int ks = 0; ks < 6; ++ks){
        int k0 = c*192 + ks*32;
        bf16x16 a = load_A(fk_sh, E_, r*16, k0, lane);
        d0 = wmma_bf16(a, load_B(Wrow, E_, 0,  k0, lane), d0);
        d1 = wmma_bf16(a, load_B(Wrow, E_, 16, k0, lane), d1);
      }
      const int n = lane & 15, half = lane >> 4;
      float* myP = Plog + c*(64*32);
      for (int rr = 0; rr < 8; ++rr){
        int row = r*16 + rr + 8*half;
        myP[row*32 + n]      = d0[rr];
        myP[row*32 + 16 + n] = d1[rr];
      }
    }
    __syncthreads();
    // exp(clip(sum of 4 K-partials)) + row denominator
    {
      const int row = tid >> 3, j0 = (tid & 7) * 4;
      float part = 0.f;
      for (int jj = 0; jj < 4; ++jj){
        int idx = row*32 + j0 + jj;
        float l = Plog[idx] + Plog[64*32 + idx] + Plog[2*64*32 + idx] + Plog[3*64*32 + idx];
        l = fminf(10.f, fmaxf(-10.f, l));
        float tv = __expf(l);
        Pb[idx] = f2bf(tv);
        part += tv;
      }
      atomicAdd(&den_sh[row], part);
    }
    __syncthreads();
    // GEMM2: acc(16 x 192) += P(16x32) @ W_tile(32 x 192)
    {
      bf16x16 a = load_A(Pb, 32, r*16, 0, lane);
      for (int nt = 0; nt < 12; ++nt)
        acc[nt] = wmma_bf16(a, load_B(Wt, 32, c*192 + nt*16, 0, lane), acc[nt]);
    }
  }
  __syncthreads();
  {
    const int n = lane & 15, half = lane >> 4;
    for (int nt = 0; nt < 12; ++nt)
      for (int rr = 0; rr < 8; ++rr){
        int row = row0g + r*16 + rr + 8*half;
        unsafeAtomicAdd(&num[(size_t)row*E_ + c*192 + nt*16 + n], acc[nt][rr]);
      }
  }
  if (tid < 64) unsafeAtomicAdd(&den[row0g + tid], den_sh[tid]);
}

// ---------------------------------------------------------------------------
// K1b: diff = e - num/den, stored TRANSPOSED [b][e][s] bf16 (so K2's LDS
// staging is a contiguous pure copy) + bias[b,e] = B_lr * sum_s diff
// ---------------------------------------------------------------------------
__global__ __launch_bounds__(256, 1)
void k_diff_bias(const float* __restrict__ e_in, const float* __restrict__ num,
                 const float* __restrict__ den, const float* __restrict__ B_lr,
                 bf16_t* __restrict__ diffT_g, float* __restrict__ bias){
  const int e0 = blockIdx.x * 256 + threadIdx.x;
  const int b  = blockIdx.y;
  const int s0 = blockIdx.z * 64;
  const float bl = B_lr[0];
  float part = 0.f;
  for (int s = s0; s < s0 + 64; ++s){
    int row = b*S_ + s;
    float dnm = den[row] + 1e-8f;
    float d = e_in[(size_t)row*E_ + e0] - num[(size_t)row*E_ + e0] / dnm;
    diffT_g[(size_t)(b*E_ + e0)*S_ + s] = f2bf(d);
    part += d;
  }
  unsafeAtomicAdd(&bias[b*E_ + e0], bl * part);
}

// ---------------------------------------------------------------------------
// K2: delta = sum_h A_lr_h * w_h (.) (attn_h @ diff); residual + LayerNorm.
// diffT staged via async pure copies; attn prefetched one head ahead.
// ---------------------------------------------------------------------------
__global__ __launch_bounds__(128, 1)
void k_attn_delta_ln(const float* __restrict__ attn, const bf16_t* __restrict__ diffT_g,
                     const float* __restrict__ f_k, const float* __restrict__ W_v,
                     const float* __restrict__ A_lr, const float* __restrict__ bias,
                     const float* __restrict__ ln_w, bf16_t* __restrict__ xn){
  extern __shared__ char smem[];
  bf16_t* diffT = (bf16_t*)smem;                  // [768][64] bf16  96KB
  bf16_t* Ash   = (bf16_t*)(smem + 64*E_*2);      // [16][64] bf16    2KB
  float*  xsh   = (float*)smem;                   // [16][768] f32 (aliases diffT)

  const int tid = threadIdx.x;
  const int lane = tid & 31, c = tid >> 5;
  const int b = blockIdx.y;
  const int row0 = blockIdx.x * 16;
  const int grow0 = b*S_ + row0;
  const int e_base = c * 192;

  const f32x8 vz = {0,0,0,0,0,0,0,0};
  f32x8 acc[12];
  for (int t = 0; t < 12; ++t) acc[t] = vz;

  for (int kc = 0; kc < S_; kc += 64){
    __syncthreads();
    for (int i8 = tid; i8 < (64*E_)/8; i8 += 128){
      int ee = i8 >> 3, ss0 = (i8 & 7) * 8;
      async_cp16(&diffT_g[(size_t)(b*E_ + ee)*S_ + kc + ss0], &diffT[ee*64 + ss0]);
    }
    wait_async();
    for (int h = 0; h < H_; ++h){
      __syncthreads();
      const float alr = A_lr[h];
      for (int i = tid; i < 16*64; i += 128){
        int m = i >> 6, ss = i & 63;
        Ash[i] = f2bf(attn[(size_t)((b*H_ + h)*S_ + row0 + m)*S_ + kc + ss] * alr);
        if (h + 1 < H_)
          __builtin_prefetch(&attn[(size_t)((b*H_ + h + 1)*S_ + row0 + m)*S_ + kc + ss], 0, 1);
      }
      __syncthreads();
      for (int nt = 0; nt < 12; ++nt){
        const float wv = W_v[h*E_ + e_base + nt*16 + (lane & 15)];
        f32x8 d = vz;
        d = wmma_bf16(load_A(Ash, 64, 0, 0,  lane), load_B(diffT, 64, e_base + nt*16, 0,  lane), d);
        d = wmma_bf16(load_A(Ash, 64, 0, 32, lane), load_B(diffT, 64, e_base + nt*16, 32, lane), d);
        for (int rr = 0; rr < 8; ++rr) acc[nt][rr] += wv * d[rr];
      }
    }
  }
  __syncthreads();
  {
    const int n = lane & 15, half = lane >> 4;
    const float inv_S = 1.0f / (float)S_;
    for (int nt = 0; nt < 12; ++nt){
      int ecol = e_base + nt*16 + n;
      float bv = bias[b*E_ + ecol];
      for (int rr = 0; rr < 8; ++rr){
        int m = rr + 8*half;
        xsh[m*E_ + ecol] = f_k[(size_t)(grow0 + m)*E_ + ecol] + (acc[nt][rr] + bv) * inv_S;
      }
    }
  }
  __syncthreads();
  for (int rr = 0; rr < 4; ++rr){
    int m = c*4 + rr;
    float s = 0.f, s2 = 0.f;
    for (int j = lane; j < E_; j += 32){
      float v = xsh[m*E_ + j];
      s += v; s2 += v*v;
    }
    for (int off = 16; off > 0; off >>= 1){
      s  += __shfl_xor(s,  off, 32);
      s2 += __shfl_xor(s2, off, 32);
    }
    float mu  = s  * (1.f/E_);
    float var = s2 * (1.f/E_) - mu*mu;
    float rs  = rsqrtf(var + LN_EPS);
    for (int j = lane; j < E_; j += 32)
      xn[(size_t)(grow0 + m)*E_ + j] = f2bf((xsh[m*E_ + j] - mu) * rs * ln_w[j]);
  }
}

// ---------------------------------------------------------------------------
// K3/K4: C = act(A @ W^T), A bf16 [M][K], W bf16 [N][K]. K-step 64; all
// staging is pure 16B async copies. 8 waves = 2x4, wave -> 16x64 of C.
// ---------------------------------------------------------------------------
template<bool GELU, bool OUTBF16>
__global__ __launch_bounds__(256, 1)
void k_gemm(const bf16_t* __restrict__ A, const bf16_t* __restrict__ W,
            void* __restrict__ out, int M, int N, int K){
  extern __shared__ char smem[];
  bf16_t* Ash = (bf16_t*)smem;        // [32][64]   4KB
  bf16_t* Bsh = Ash + 32*64;          // [256][64] 32KB
  const int tid = threadIdx.x, lane = tid & 31, wave = tid >> 5;
  const int wy = wave >> 2, wx = wave & 3;
  const int mrow = blockIdx.x * 32;
  const int ncol = blockIdx.y * 256;

  const f32x8 vz = {0,0,0,0,0,0,0,0};
  f32x8 acc[4] = {vz, vz, vz, vz};

  for (int k0 = 0; k0 < K; k0 += 64){
    __syncthreads();
    for (int i8 = tid; i8 < 32*8; i8 += 256){
      int m = i8 >> 3, kk = (i8 & 7) * 8;
      async_cp16(&A[(size_t)(mrow + m)*K + k0 + kk], &Ash[m*64 + kk]);
    }
    for (int i8 = tid; i8 < 256*8; i8 += 256){
      int nn = i8 >> 3, kk = (i8 & 7) * 8;
      async_cp16(&W[(size_t)(ncol + nn)*K + k0 + kk], &Bsh[nn*64 + kk]);
    }
    wait_async();
    __syncthreads();
    bf16x16 a0 = load_A(Ash, 64, wy*16, 0,  lane);
    bf16x16 a1 = load_A(Ash, 64, wy*16, 32, lane);
    for (int nt = 0; nt < 4; ++nt){
      acc[nt] = wmma_bf16(a0, load_B(Bsh, 64, wx*64 + nt*16, 0,  lane), acc[nt]);
      acc[nt] = wmma_bf16(a1, load_B(Bsh, 64, wx*64 + nt*16, 32, lane), acc[nt]);
    }
  }
  const int n = lane & 15, half = lane >> 4;
  for (int nt = 0; nt < 4; ++nt){
    for (int rr = 0; rr < 8; ++rr){
      int row = mrow + wy*16 + rr + 8*half;
      int col = ncol + wx*64 + nt*16 + n;
      float v = acc[nt][rr];
      if (GELU) v = 0.5f * v * (1.f + erff(v * 0.70710678118654752f));
      if (OUTBF16) ((bf16_t*)out)[(size_t)row*N + col] = f2bf(v);
      else         ((float*)out)[(size_t)row*N + col] = v;
    }
  }
}

// ---------------------------------------------------------------------------
extern "C" void kernel_launch(void* const* d_in, const int* in_sizes, int n_in,
                              void* d_out, int out_size, void* d_ws, size_t ws_size,
                              hipStream_t stream) {
  (void)in_sizes; (void)n_in; (void)out_size; (void)ws_size;
  const float* f_k  = (const float*)d_in[0];
  const float* attn = (const float*)d_in[1];
  const float* e_in = (const float*)d_in[2];
  const float* W_e  = (const float*)d_in[3];
  const float* W_v  = (const float*)d_in[4];
  const float* A_lr = (const float*)d_in[5];
  const float* B_lr = (const float*)d_in[6];
  const float* ln_w = (const float*)d_in[7];
  const float* W1   = (const float*)d_in[8];
  const float* W2   = (const float*)d_in[9];
  float* out = (float*)d_out;

  char* ws = (char*)d_ws;
  // region A: num (12.6MB f32, dead after K1b) reused as hbuf (25.2MB bf16)
  float*  num   = (float*)ws;
  bf16_t* hbuf  = (bf16_t*)ws;
  size_t off    = (size_t)ROWS_ * DFF_ * 2;                     // 25165824
  float*  den   = (float*)(ws + off);  off += (size_t)ROWS_ * 4;
  float*  bias  = (float*)(ws + off);  off += (size_t)B_ * E_ * 4;
  bf16_t* diffT = (bf16_t*)(ws + off); off += (size_t)ROWS_ * E_ * 2;
  bf16_t* xn    = (bf16_t*)(ws + off); off += (size_t)ROWS_ * E_ * 2;
  bf16_t* We_bf = (bf16_t*)(ws + off); off += (size_t)V_ * E_ * 2;
  bf16_t* W1_bf = (bf16_t*)(ws + off); off += (size_t)DFF_ * E_ * 2;
  bf16_t* W2_bf = (bf16_t*)(ws + off); off += (size_t)E_ * DFF_ * 2;

  (void)hipMemsetAsync(num,  0, (size_t)ROWS_ * E_ * 4, stream);
  (void)hipMemsetAsync(den,  0, (size_t)ROWS_ * 4, stream);
  (void)hipMemsetAsync(bias, 0, (size_t)B_ * E_ * 4, stream);

  k_cvt_bf16<<<(V_*E_/8 + 255)/256,   256, 0, stream>>>(W_e, We_bf, V_*E_/8);
  k_cvt_bf16<<<(DFF_*E_/8 + 255)/256, 256, 0, stream>>>(W1,  W1_bf, DFF_*E_/8);
  k_cvt_bf16<<<(DFF_*E_/8 + 255)/256, 256, 0, stream>>>(W2,  W2_bf, DFF_*E_/8);

  k_vocab_expect<<<dim3(ROWS_/64, 4), 512, 233728, stream>>>(f_k, We_bf, num, den);
  k_diff_bias<<<dim3(E_/256, B_, S_/64), 256, 0, stream>>>(e_in, num, den, B_lr, diffT, bias);
  k_attn_delta_ln<<<dim3(S_/16, B_), 128, 100352, stream>>>(attn, diffT, f_k, W_v, A_lr, bias, ln_w, xn);
  k_gemm<true,  true ><<<dim3(ROWS_/32, DFF_/256), 256, 36864, stream>>>(xn,   W1_bf, (void*)hbuf, ROWS_, DFF_, E_);
  k_gemm<false, false><<<dim3(ROWS_/32, E_/256),   256, 36864, stream>>>(hbuf, W2_bf, (void*)out,  ROWS_, E_,  DFF_);
}
